// ChebBenX_71159018160655
// MI455X (gfx1250) — matrix operand order repaired and checked by reading, not compiled, not run on hardware
//
#include <hip/hip_runtime.h>
#include <hip/hip_bf16.h>
#include <stdint.h>

typedef __attribute__((ext_vector_type(16))) __bf16 v16bf;
typedef __attribute__((ext_vector_type(8)))  float  v8f;

#define N_NODES 50000
#define N_EDGES 800000

static inline int cdiv(int a, int b) { return (a + b - 1) / b; }

// ---------------------------------------------------------------- utilities

__global__ void fill_zero_kernel(float* __restrict__ p, int n) {
    int i = blockIdx.x * blockDim.x + threadIdx.x;
    if (i < n) p[i] = 0.0f;
}

__global__ void negcopy_kernel(const float* __restrict__ in, float* __restrict__ out, int n) {
    int i = blockIdx.x * blockDim.x + threadIdx.x;
    if (i < n) out[i] = -in[i];
}

// deg[src] += 1  (edge_index is int64: row 0 = src, row 1 = dst)
__global__ void deg_kernel(const long long* __restrict__ ei, float* __restrict__ deg) {
    int e = blockIdx.x * blockDim.x + threadIdx.x;
    if (e < N_EDGES) unsafeAtomicAdd(&deg[(int)ei[e]], 1.0f);
}

// dis[i] = deg>0 ? rsqrt(max(deg,1)) : 0   (in place)
__global__ void dis_kernel(float* __restrict__ d) {
    int i = blockIdx.x * blockDim.x + threadIdx.x;
    if (i < N_NODES) {
        float v = d[i];
        d[i] = (v > 0.0f) ? rsqrtf(fmaxf(v, 1.0f)) : 0.0f;
    }
}

// One pass: narrow the int64 edge index to int32 and compute the symmetric
// normalization weight. The 6 scatter passes then read only 12 B/edge of
// edge metadata instead of 20 B.
__global__ void edge_prep_kernel(const long long* __restrict__ ei, const float* __restrict__ dis,
                                 int* __restrict__ src32, int* __restrict__ dst32,
                                 float* __restrict__ w) {
    int e = blockIdx.x * blockDim.x + threadIdx.x;
    if (e < N_EDGES) {
        int s = (int)ei[e];
        int d = (int)ei[N_EDGES + e];
        src32[e] = s;
        dst32[e] = d;
        w[e] = dis[s] * dis[d];
    }
}

// out[dst] += coef * w[e] * in[src]  over F features, float4 per thread.
// Consecutive threads cover consecutive feature chunks of the same edge ->
// coalesced global_load_b128 gathers; scatter via hardware global_atomic_add_f32
// (resolves in L2: all feature matrices are L2-resident).
template <int F>
__global__ void lap_scatter_kernel(const int* __restrict__ src32, const int* __restrict__ dst32,
                                   const float* __restrict__ w,
                                   const float* __restrict__ in, float* __restrict__ out,
                                   float coef) {
    constexpr int CH = F / 4;
    int t = blockIdx.x * blockDim.x + threadIdx.x;
    int e = t / CH;
    if (e >= N_EDGES) return;
    int c = (t % CH) * 4;
    int s = src32[e];
    int d = dst32[e];
    float cw = coef * w[e];
    const float4 v = *(const float4*)(in + (size_t)s * F + c);
    float* o = out + (size_t)d * F + c;
    unsafeAtomicAdd(o + 0, cw * v.x);
    unsafeAtomicAdd(o + 1, cw * v.y);
    unsafeAtomicAdd(o + 2, cw * v.z);
    unsafeAtomicAdd(o + 3, cw * v.w);
}

// ---------------------------------------------------------------- WMMA GEMM
// out[N, Fo] = relu?( T0@W[0] + T1@W[1] + T2@W[2] + bias )
// Block = 128 threads = 4 wave32. Each wave owns one 16x16 output tile.
// Fo=64 -> 4 col tiles, 1 row tile (16 rows/block); Fo=16 -> 1 col tile,
// 4 row tiles (64 rows/block). A is staged fp32->bf16 through LDS; fragments
// follow ISA 7.12.2 wave32 16-bit layouts; accumulate fp32 in v8f across all
// 3 Chebyshev orders (3*Fi/32 v_wmma_f32_16x16x32_bf16 per tile).
template <int Fi, int Fo, bool RELU>
__global__ void __launch_bounds__(128)
cheb_gemm_kernel(const float* __restrict__ T0, const float* __restrict__ T1,
                 const float* __restrict__ T2, const float* __restrict__ W,
                 const float* __restrict__ bias, float* __restrict__ out) {
    constexpr int nCT  = Fo / 16;
    constexpr int nRT  = 4 / nCT;
    constexpr int ROWS = 16 * nRT;
    __shared__ __bf16 As[ROWS * Fi];

    const int lane = threadIdx.x & 31;
    const int wave = threadIdx.x >> 5;
    const int ct   = wave % nCT;
    const int rt   = wave / nCT;
    const int hi   = lane >> 4;    // half-wave select
    const int lr   = lane & 15;
    const int rowBase = blockIdx.x * ROWS;

    v8f acc = {};

    const float* Ts[3] = {T0, T1, T2};
    for (int k = 0; k < 3; ++k) {
        __syncthreads();
        const float* __restrict__ T = Ts[k];
        for (int idx = threadIdx.x; idx < ROWS * Fi; idx += 128) {
            int r  = idx / Fi;
            int c  = idx % Fi;
            int gr = rowBase + r;
            float v = (gr < N_NODES) ? T[(size_t)gr * Fi + c] : 0.0f;
            As[idx] = (__bf16)v;
        }
        __syncthreads();

        const float* __restrict__ Wk = W + (size_t)k * Fi * Fo;
        for (int kk = 0; kk < Fi; kk += 32) {
            v16bf a, b;
            const __bf16* arow = &As[(rt * 16 + lr) * Fi + kk];
#pragma unroll
            for (int e2 = 0; e2 < 16; ++e2) {
                // A (16x32 bf16): lane<16 holds K {0..7,16..23}, lane>=16 holds K {8..15,24..31}
                int ka = e2 + hi * 8 + ((e2 >= 8) ? 8 : 0);
                a[e2] = arow[ka];
                // B (32x16 bf16): lanes 0-15 hold K {0..7,16..23} (col = lr), lanes 16-31 hold K {8..15,24..31}
                int kb = e2 + hi * 8 + ((e2 >= 8) ? 8 : 0);
                b[e2] = (__bf16)Wk[(size_t)(kk + kb) * Fo + ct * 16 + lr];
            }
            acc = __builtin_amdgcn_wmma_f32_16x16x32_bf16(
                false, a, false, b, (short)0, acc, false, false);
        }
    }

    // C/D layout: VGPR r -> M = hi*8 + r, N = lane%16
    const int col = ct * 16 + lr;
    const float bv = bias[col];
    float* op = out + (size_t)(rowBase + rt * 16 + hi * 8) * Fo + col;
    if (rowBase + ROWS <= N_NODES) {
        // block-uniform fast path: no exec-mask manipulation, 8 plain stores
#pragma unroll
        for (int r = 0; r < 8; ++r) {
            float v = acc[r] + bv;
            if (RELU) v = fmaxf(v, 0.0f);
            op[(size_t)r * Fo] = v;
        }
    } else {
#pragma unroll
        for (int r = 0; r < 8; ++r) {
            int row = rowBase + rt * 16 + hi * 8 + r;
            if (row < N_NODES) {
                float v = acc[r] + bv;
                if (RELU) v = fmaxf(v, 0.0f);
                op[(size_t)r * Fo] = v;
            }
        }
    }
}

// ---------------------------------------------------------------- launch

static void run_layer(const int* src32, const int* dst32, const float* wE,
                      const float* in, float* t1, float* t2,
                      const float* W, const float* bias, float* out, int Fi, int Fo,
                      hipStream_t stream) {
    const int nF = N_NODES * Fi;
    fill_zero_kernel<<<cdiv(nF, 256), 256, 0, stream>>>(t1, nF);
    negcopy_kernel<<<cdiv(nF, 256), 256, 0, stream>>>(in, t2, nF);
    const int sThreads = N_EDGES * (Fi / 4);
    if (Fi == 128) {
        lap_scatter_kernel<128><<<cdiv(sThreads, 256), 256, 0, stream>>>(src32, dst32, wE, in, t1, -1.0f);
        lap_scatter_kernel<128><<<cdiv(sThreads, 256), 256, 0, stream>>>(src32, dst32, wE, t1, t2, -2.0f);
    } else {
        lap_scatter_kernel<64><<<cdiv(sThreads, 256), 256, 0, stream>>>(src32, dst32, wE, in, t1, -1.0f);
        lap_scatter_kernel<64><<<cdiv(sThreads, 256), 256, 0, stream>>>(src32, dst32, wE, t1, t2, -2.0f);
    }
    if (Fi == 128 && Fo == 64) {
        cheb_gemm_kernel<128, 64, true><<<cdiv(N_NODES, 16), 128, 0, stream>>>(in, t1, t2, W, bias, out);
    } else if (Fi == 64 && Fo == 64) {
        cheb_gemm_kernel<64, 64, true><<<cdiv(N_NODES, 16), 128, 0, stream>>>(in, t1, t2, W, bias, out);
    } else { // 64 -> 16, no relu
        cheb_gemm_kernel<64, 16, false><<<cdiv(N_NODES, 64), 128, 0, stream>>>(in, t1, t2, W, bias, out);
    }
}

extern "C" void kernel_launch(void* const* d_in, const int* in_sizes, int n_in,
                              void* d_out, int out_size, void* d_ws, size_t ws_size,
                              hipStream_t stream) {
    (void)in_sizes; (void)n_in; (void)out_size; (void)ws_size;

    const float*     x  = (const float*)d_in[0];         // [N,128]
    const long long* ei = (const long long*)d_in[1];     // [2,E] int64
    const float*     W1 = (const float*)d_in[2];         // [3,128,64]
    const float*     b1 = (const float*)d_in[3];
    const float*     Wm = (const float*)d_in[4];         // [3,64,64]
    const float*     bm = (const float*)d_in[5];
    const float*     W2 = (const float*)d_in[6];         // [3,64,16]
    const float*     b2 = (const float*)d_in[7];
    float* out = (float*)d_out;                          // [N,16]

    float* ws    = (float*)d_ws;
    float* dis   = ws;                                   // N floats
    float* wE    = dis + N_NODES;                        // E floats
    int*   src32 = (int*)(wE + N_EDGES);                 // E ints
    int*   dst32 = src32 + N_EDGES;                      // E ints
    float* t1    = (float*)(dst32 + N_EDGES);            // N*128 (reused at width 64)
    float* t2    = t1 + (size_t)N_NODES * 128;           // N*128
    float* h1    = t2 + (size_t)N_NODES * 128;           // N*64
    float* h2    = h1 + (size_t)N_NODES * 64;            // N*64

    // normalization weights
    fill_zero_kernel<<<cdiv(N_NODES, 256), 256, 0, stream>>>(dis, N_NODES);
    deg_kernel<<<cdiv(N_EDGES, 256), 256, 0, stream>>>(ei, dis);
    dis_kernel<<<cdiv(N_NODES, 256), 256, 0, stream>>>(dis);
    edge_prep_kernel<<<cdiv(N_EDGES, 256), 256, 0, stream>>>(ei, dis, src32, dst32, wE);

    run_layer(src32, dst32, wE, x,  t1, t2, W1, b1, h1, 128, 64, stream);
    run_layer(src32, dst32, wE, h1, t1, t2, Wm, bm, h2,  64, 64, stream);
    run_layer(src32, dst32, wE, h2, t1, t2, W2, b2, out, 64, 16, stream);
}
